// RelPosMultiHeadSelfAttention_22728966930580
// MI455X (gfx1250) — compile-verified, matrix operand order checked by reading, not run on hardware
//
#include <hip/hip_runtime.h>
#include <hip/hip_bf16.h>

typedef __bf16 bf16_t;
typedef __attribute__((ext_vector_type(16))) bf16_t v16bf;
typedef __attribute__((ext_vector_type(8)))  bf16_t v8bf;
typedef __attribute__((ext_vector_type(8)))  float  v8f;

#define S_LEN 512
#define M_LEN 512
#define K_LEN 1024
#define B_SZ  8
#define NHEAD 16
#define HDIM  64
#define HDN   1024
#define TOTAL_LEN 1536
#define SOFT_SCALE 0.125f   // 1/sqrt(64)

// ---------------- WMMA helpers (layouts per CDNA5 ISA 7.12.2) ----

// A/B fragment from a row-major bf16 matrix: row = row0 + lane%16,
// elements 0..7 -> k0 + (lane/16)*8 + e ; elements 8..15 -> +16.
__device__ __forceinline__ v16bf load_frag(const bf16_t* base, int ld,
                                           int row0, int k0, int lane) {
  int r = row0 + (lane & 15);
  int k = k0 + ((lane >> 4) << 3);
  const bf16_t* p = base + (size_t)r * ld + k;
  v8bf lo = *(const v8bf*)(p);
  v8bf hi = *(const v8bf*)(p + 16);
  v16bf a;
#pragma unroll
  for (int e = 0; e < 8; ++e) { a[e] = lo[e]; a[e + 8] = hi[e]; }
  return a;
}

__device__ __forceinline__ v8f wmma_bf16(v16bf a, v16bf b, v8f c) {
  return __builtin_amdgcn_wmma_f32_16x16x32_bf16(false, a, false, b,
                                                 (short)0, c, false, false);
}

// Register-blocked macro-tile: one wave computes 32 rows x 64 cols.
// acc[h*4+t] = tile (m0+16h, n0+16t). 8 WMMAs per 6 fragment loads per k-step.
__device__ __forceinline__ void gemm_32x64(const bf16_t* A, int lda,
                                           const bf16_t* Bt, int ldb,
                                           int m0, int n0, int kdim, int lane,
                                           v8f acc[8]) {
#pragma unroll 1
  for (int k = 0; k < kdim; k += 32) {
    v16bf a0 = load_frag(A, lda, m0,      k, lane);
    v16bf a1 = load_frag(A, lda, m0 + 16, k, lane);
#pragma unroll
    for (int t = 0; t < 4; ++t) {
      v16bf b = load_frag(Bt, ldb, n0 + t * 16, k, lane);
      acc[t]     = wmma_bf16(a0, b, acc[t]);
      acc[4 + t] = wmma_bf16(a1, b, acc[4 + t]);
    }
    if (k + 64 < kdim) {   // stream next k-panel into cache (global_prefetch_b8)
      __builtin_prefetch(A  + (size_t)(m0 + (lane & 31)) * lda + k + 64, 0, 1);
      __builtin_prefetch(Bt + (size_t)(n0 + (lane & 31)) * ldb + k + 64, 0, 1);
      __builtin_prefetch(Bt + (size_t)(n0 + 32 + (lane & 31)) * ldb + k + 64, 0, 1);
    }
  }
}

__device__ __forceinline__ void zero_acc(v8f acc[8]) {
#pragma unroll
  for (int i = 0; i < 8; ++i)
    acc[i] = (v8f){0.f, 0.f, 0.f, 0.f, 0.f, 0.f, 0.f, 0.f};
}

// ---------------- prep kernels ---------------------------------------------

__global__ void cvt_f32_bf16(const float* __restrict__ in,
                             bf16_t* __restrict__ out, int n) {
  int i = blockIdx.x * blockDim.x + threadIdx.x;
  if (i < n) out[i] = (bf16_t)in[i];
}

// KV flat (B*K, H): row r -> b=r/K, j=r%K; j<M from memories, else x.
__global__ void build_kv(const float* __restrict__ x,
                         const float* __restrict__ memories,
                         const int* __restrict__ lp, const int* __restrict__ sp,
                         bf16_t* __restrict__ KV) {
  int idx = blockIdx.x * 256 + threadIdx.x;           // < 8192*1024
  int r = idx >> 10, h = idx & 1023;
  int b = r >> 10;                                    // r / K_LEN
  int j = r & 1023;                                   // r % K_LEN
  float val;
  if (j < M_LEN) {
    size_t off = (((size_t)(lp[0] * B_SZ + b)) * TOTAL_LEN +
                  (size_t)(sp[0] * M_LEN + j)) * HDN + h;
    val = memories[off];
  } else {
    val = x[((size_t)b * S_LEN + (j - M_LEN)) * HDN + h];
  }
  KV[idx] = (bf16_t)val;
}

// WT[n][k] = bf16(W[k][n]), 1024x1024, LDS tiled
__global__ void transpose_w(const float* __restrict__ W,
                            bf16_t* __restrict__ WT) {
  __shared__ float t[16][17];
  int bx = blockIdx.x * 16, by = blockIdx.y * 16;
  int tx = threadIdx.x & 15, ty = threadIdx.x >> 4;   // blockDim = 256
  t[ty][tx] = W[(size_t)(by + ty) * HDN + bx + tx];
  __syncthreads();
  WT[(size_t)(bx + ty) * HDN + by + tx] = (bf16_t)t[tx][ty];
}

// ---------------- projection GEMMs (4 waves/block, 32x64 tile/wave) ---------

__global__ void proj_q(const bf16_t* __restrict__ A, const bf16_t* __restrict__ Bt,
                       const float* __restrict__ u, const float* __restrict__ v,
                       bf16_t* __restrict__ qu, bf16_t* __restrict__ qv) {
  int lane = threadIdx.x & 31, wave = threadIdx.x >> 5;
  int wt = blockIdx.x * 4 + wave;                     // 128 x 16 wave-tiles
  int m0 = (wt >> 4) * 32, n0 = (wt & 15) * 64;
  v8f acc[8]; zero_acc(acc);
  gemm_32x64(A, HDN, Bt, HDN, m0, n0, HDN, lane, acc);
  int nn = lane & 15, mm = (lane >> 4) * 8;
#pragma unroll
  for (int h2 = 0; h2 < 2; ++h2)
#pragma unroll
    for (int t = 0; t < 4; ++t) {
      v8f c = acc[h2 * 4 + t];
#pragma unroll
      for (int r = 0; r < 8; ++r) {
        int row = m0 + h2 * 16 + mm + r;              // flat (S*B) row
        int h   = n0 + t * 16 + nn;
        int b = row & 7, i = row >> 3;
        int nh = h >> 6, d = h & 63;
        size_t o = (((size_t)(b * NHEAD + nh)) * S_LEN + i) * HDIM + d;
        qu[o] = (bf16_t)(c[r] + u[nh * HDIM + d]);
        qv[o] = (bf16_t)(c[r] + v[nh * HDIM + d]);
      }
    }
}

__global__ void proj_k(const bf16_t* __restrict__ A, const bf16_t* __restrict__ Bt,
                       bf16_t* __restrict__ Kb) {
  int lane = threadIdx.x & 31, wave = threadIdx.x >> 5;
  int wt = blockIdx.x * 4 + wave;                     // 256 x 16 wave-tiles
  int m0 = (wt >> 4) * 32, n0 = (wt & 15) * 64;
  v8f acc[8]; zero_acc(acc);
  gemm_32x64(A, HDN, Bt, HDN, m0, n0, HDN, lane, acc);
  int nn = lane & 15, mm = (lane >> 4) * 8;
#pragma unroll
  for (int h2 = 0; h2 < 2; ++h2)
#pragma unroll
    for (int t = 0; t < 4; ++t) {
      v8f c = acc[h2 * 4 + t];
#pragma unroll
      for (int r = 0; r < 8; ++r) {
        int row = m0 + h2 * 16 + mm + r;              // flat (K*B) row
        int h   = n0 + t * 16 + nn;
        int b = row & 7, j = row >> 3;
        int nh = h >> 6, d = h & 63;
        Kb[(((size_t)(b * NHEAD + nh)) * K_LEN + j) * HDIM + d] = (bf16_t)c[r];
      }
    }
}

__global__ void proj_v(const bf16_t* __restrict__ A, const bf16_t* __restrict__ Bt,
                       bf16_t* __restrict__ Vt) {
  int lane = threadIdx.x & 31, wave = threadIdx.x >> 5;
  int wt = blockIdx.x * 4 + wave;
  int m0 = (wt >> 4) * 32, n0 = (wt & 15) * 64;
  v8f acc[8]; zero_acc(acc);
  gemm_32x64(A, HDN, Bt, HDN, m0, n0, HDN, lane, acc);
  int nn = lane & 15, mm = (lane >> 4) * 8;
#pragma unroll
  for (int h2 = 0; h2 < 2; ++h2)
#pragma unroll
    for (int t = 0; t < 4; ++t) {
      v8f c = acc[h2 * 4 + t];
#pragma unroll
      for (int r = 0; r < 8; ++r) {
        int row = m0 + h2 * 16 + mm + r;
        int h   = n0 + t * 16 + nn;
        int b = row & 7, j = row >> 3;
        int nh = h >> 6, d = h & 63;
        // transposed store: Vt[b][n][d][j]
        Vt[(((size_t)(b * NHEAD + nh)) * HDIM + d) * K_LEN + j] = (bf16_t)c[r];
      }
    }
}

__global__ void proj_kr(const bf16_t* __restrict__ A, const bf16_t* __restrict__ Bt,
                        bf16_t* __restrict__ kr) {
  int lane = threadIdx.x & 31, wave = threadIdx.x >> 5;
  int wt = blockIdx.x * 4 + wave;                     // 32 x 16 wave-tiles
  int m0 = (wt >> 4) * 32, n0 = (wt & 15) * 64;
  v8f acc[8]; zero_acc(acc);
  gemm_32x64(A, HDN, Bt, HDN, m0, n0, HDN, lane, acc);
  int nn = lane & 15, mm = (lane >> 4) * 8;
#pragma unroll
  for (int h2 = 0; h2 < 2; ++h2)
#pragma unroll
    for (int t = 0; t < 4; ++t) {
      v8f c = acc[h2 * 4 + t];
#pragma unroll
      for (int r = 0; r < 8; ++r) {
        int j = m0 + h2 * 16 + mm + r;                // no batch scramble
        int h = n0 + t * 16 + nn;
        int nh = h >> 6, d = h & 63;
        kr[((size_t)nh * K_LEN + j) * HDIM + d] = (bf16_t)c[r];
      }
    }
}

// ---------------- fused attention ------------------------------------------
// grid (B*NH, S/16), block 128 (4 waves), 64KB dynamic LDS: sc[16][1024]

__global__ void attention_kernel(const bf16_t* __restrict__ qu,
                                 const bf16_t* __restrict__ qv,
                                 const bf16_t* __restrict__ Kb,
                                 const bf16_t* __restrict__ Vt,
                                 const bf16_t* __restrict__ krb,
                                 bf16_t* __restrict__ AO) {
  extern __shared__ float sc[];                       // 16 * 1024 f32
  int bn = blockIdx.x;
  int b = bn / NHEAD, n = bn % NHEAD;
  int i0 = blockIdx.y * 16;
  int lane = threadIdx.x & 31, wave = threadIdx.x >> 5;

  const bf16_t* qu_bn = qu  + (size_t)bn * S_LEN * HDIM;
  const bf16_t* qv_bn = qv  + (size_t)bn * S_LEN * HDIM;
  const bf16_t* K_bn  = Kb  + (size_t)bn * K_LEN * HDIM;
  const bf16_t* V_bn  = Vt  + (size_t)bn * HDIM * K_LEN;
  const bf16_t* kr_n  = krb + (size_t)n  * K_LEN * HDIM;

  for (int t = threadIdx.x; t < 16 * K_LEN; t += blockDim.x) sc[t] = 0.f;
  __syncthreads();

  v16bf aqv0 = load_frag(qv_bn, HDIM, i0, 0, lane);
  v16bf aqv1 = load_frag(qv_bn, HDIM, i0, 32, lane);
  v16bf aqu0 = load_frag(qu_bn, HDIM, i0, 0, lane);
  v16bf aqu1 = load_frag(qu_bn, HDIM, i0, 32, lane);
  int nn = lane & 15, mm = (lane >> 4) * 8;

  // bd via WMMA, scattered through rel_shift: j = jj - (S-1-i)
  for (int jt = wave; jt < 64; jt += 4) {
    v8f c = {0.f, 0.f, 0.f, 0.f, 0.f, 0.f, 0.f, 0.f};
    v16bf b0 = load_frag(kr_n, HDIM, jt * 16, 0, lane);
    v16bf b1 = load_frag(kr_n, HDIM, jt * 16, 32, lane);
    c = wmma_bf16(aqv0, b0, c);
    c = wmma_bf16(aqv1, b1, c);
#pragma unroll
    for (int r = 0; r < 8; ++r) {
      int il = mm + r;
      int i  = i0 + il;
      int jj = jt * 16 + nn;
      int j  = jj - (S_LEN - 1 - i);
      if (j >= 0 && j < K_LEN) sc[il * K_LEN + j] = c[r];
    }
  }
  __syncthreads();

  // ac via WMMA, accumulate on top of bd
  for (int jt = wave; jt < 64; jt += 4) {
    v8f c = {0.f, 0.f, 0.f, 0.f, 0.f, 0.f, 0.f, 0.f};
    v16bf b0 = load_frag(K_bn, HDIM, jt * 16, 0, lane);
    v16bf b1 = load_frag(K_bn, HDIM, jt * 16, 32, lane);
    c = wmma_bf16(aqu0, b0, c);
    c = wmma_bf16(aqu1, b1, c);
#pragma unroll
    for (int r = 0; r < 8; ++r)
      sc[(mm + r) * K_LEN + jt * 16 + nn] += c[r];
  }
  __syncthreads();

  // softmax (scaled), wave handles 4 rows, lanes stride columns
  for (int rr = wave; rr < 16; rr += 4) {
    float* row = sc + rr * K_LEN;
    float mx = -3.0e38f;
    for (int j = lane; j < K_LEN; j += 32)
      mx = fmaxf(mx, row[j] * SOFT_SCALE);
    for (int o = 16; o >= 1; o >>= 1) mx = fmaxf(mx, __shfl_xor(mx, o, 32));
    float sm = 0.f;
    for (int j = lane; j < K_LEN; j += 32) {
      float e = __expf(row[j] * SOFT_SCALE - mx);
      row[j] = e;
      sm += e;
    }
    for (int o = 16; o >= 1; o >>= 1) sm += __shfl_xor(sm, o, 32);
    float inv = 1.0f / sm;
    for (int j = lane; j < K_LEN; j += 32) row[j] *= inv;
  }
  __syncthreads();

  // out = P @ V ; wave w owns d-tile w (d in [16w,16w+16))
  {
    int dt = wave;
    v8f c = {0.f, 0.f, 0.f, 0.f, 0.f, 0.f, 0.f, 0.f};
    for (int kt = 0; kt < K_LEN; kt += 32) {
      v16bf a;
      int m  = lane & 15;
      int kb = (lane >> 4) * 8;
#pragma unroll
      for (int e = 0; e < 8; ++e) {
        a[e]     = (bf16_t)sc[m * K_LEN + kt + kb + e];
        a[e + 8] = (bf16_t)sc[m * K_LEN + kt + 16 + kb + e];
      }
      v16bf bf = load_frag(V_bn, K_LEN, dt * 16, kt, lane);
      c = wmma_bf16(a, bf, c);
    }
#pragma unroll
    for (int r = 0; r < 8; ++r) {
      int i = i0 + mm + r;
      int d = dt * 16 + nn;
      size_t q = (size_t)i * B_SZ + b;                // flat (S*B) row
      AO[q * HDN + n * HDIM + d] = (bf16_t)c[r];
    }
  }
}

// ---------------- epilogue --------------------------------------------------

__global__ void copy_mem(const float4* __restrict__ src, float4* __restrict__ dst) {
  int i = blockIdx.x * 256 + threadIdx.x;             // 6291456 float4
  dst[i] = src[i];
}

__global__ void final_proj(const bf16_t* __restrict__ AO,
                           const bf16_t* __restrict__ WoT,
                           const float* __restrict__ bo,
                           const int* __restrict__ lp, const int* __restrict__ sp,
                           float* __restrict__ out) {
  int lane = threadIdx.x & 31, wave = threadIdx.x >> 5;
  int wt = blockIdx.x * 4 + wave;                     // 128 x 16 wave-tiles
  int m0 = (wt >> 4) * 32, n0 = (wt & 15) * 64;
  v8f acc[8]; zero_acc(acc);
  gemm_32x64(AO, HDN, WoT, HDN, m0, n0, HDN, lane, acc);
  int l = lp[0], sg = sp[0];
  float* mem = out + (size_t)B_SZ * S_LEN * HDN;
  int nn = lane & 15, mm = (lane >> 4) * 8;
#pragma unroll
  for (int h2 = 0; h2 < 2; ++h2)
#pragma unroll
    for (int t = 0; t < 4; ++t) {
      v8f c = acc[h2 * 4 + t];
#pragma unroll
      for (int r = 0; r < 8; ++r) {
        int row = m0 + h2 * 16 + mm + r;              // (B,S,H) flat row
        int h   = n0 + t * 16 + nn;
        float val = c[r] + bo[h];
        out[(size_t)row * HDN + h] = val;
        int bb = row >> 9;                            // row / 512
        int ss = row & 511;
        size_t tt = (size_t)sg * S_LEN + M_LEN + ss;
        mem[(((size_t)l * B_SZ + bb) * TOTAL_LEN + tt) * HDN + h] = val;
      }
    }
}

// ---------------- launcher --------------------------------------------------

extern "C" void kernel_launch(void* const* d_in, const int* in_sizes, int n_in,
                              void* d_out, int out_size, void* d_ws, size_t ws_size,
                              hipStream_t stream) {
  (void)in_sizes; (void)n_in; (void)out_size; (void)ws_size;
  const float* x    = (const float*)d_in[0];
  const float* rel  = (const float*)d_in[1];
  const float* mems = (const float*)d_in[3];
  const float* Wq   = (const float*)d_in[4];
  const float* Wke  = (const float*)d_in[5];
  const float* Wv   = (const float*)d_in[6];
  const float* Wkr  = (const float*)d_in[7];
  const float* u    = (const float*)d_in[8];
  const float* v    = (const float*)d_in[9];
  const float* Wo   = (const float*)d_in[10];
  const float* bo   = (const float*)d_in[11];
  const int*   lp   = (const int*)d_in[12];
  const int*   sp   = (const int*)d_in[13];
  float* out = (float*)d_out;

  // workspace carve-out (256B aligned)
  char* w = (char*)d_ws;
  size_t off = 0;
  auto carve = [&](size_t bytes) {
    char* p = w + off;
    off += (bytes + 255) & ~(size_t)255;
    return p;
  };
  bf16_t* Xbf   = (bf16_t*)carve((size_t)4096 * 1024 * 2);
  bf16_t* KVbf  = (bf16_t*)carve((size_t)8192 * 1024 * 2);
  bf16_t* relbf = (bf16_t*)carve((size_t)1024 * 1024 * 2);
  bf16_t* WqT   = (bf16_t*)carve((size_t)1024 * 1024 * 2);
  bf16_t* WkeT  = (bf16_t*)carve((size_t)1024 * 1024 * 2);
  bf16_t* WvT   = (bf16_t*)carve((size_t)1024 * 1024 * 2);
  bf16_t* WkrT  = (bf16_t*)carve((size_t)1024 * 1024 * 2);
  bf16_t* WoT   = (bf16_t*)carve((size_t)1024 * 1024 * 2);
  bf16_t* qu    = (bf16_t*)carve((size_t)B_SZ * NHEAD * S_LEN * HDIM * 2);
  bf16_t* qvb   = (bf16_t*)carve((size_t)B_SZ * NHEAD * S_LEN * HDIM * 2);
  bf16_t* Kb    = (bf16_t*)carve((size_t)B_SZ * NHEAD * K_LEN * HDIM * 2);
  bf16_t* Vt    = (bf16_t*)carve((size_t)B_SZ * NHEAD * HDIM * K_LEN * 2);
  bf16_t* krb   = (bf16_t*)carve((size_t)NHEAD * K_LEN * HDIM * 2);
  bf16_t* AO    = (bf16_t*)carve((size_t)4096 * 1024 * 2);

  // 1. conversions / kv build
  cvt_f32_bf16<<<(4194304 + 255) / 256, 256, 0, stream>>>(x, Xbf, 4194304);
  cvt_f32_bf16<<<(1048576 + 255) / 256, 256, 0, stream>>>(rel, relbf, 1048576);
  build_kv<<<8388608 / 256, 256, 0, stream>>>(x, mems, lp, sp, KVbf);

  // 2. weight transposes
  dim3 tg(64, 64);
  transpose_w<<<tg, 256, 0, stream>>>(Wq,  WqT);
  transpose_w<<<tg, 256, 0, stream>>>(Wke, WkeT);
  transpose_w<<<tg, 256, 0, stream>>>(Wv,  WvT);
  transpose_w<<<tg, 256, 0, stream>>>(Wkr, WkrT);
  transpose_w<<<tg, 256, 0, stream>>>(Wo,  WoT);

  // 3. projections (WMMA, 32x64 macro-tiles)
  proj_q <<<512,  128, 0, stream>>>(Xbf,  WqT,  u, v, qu, qvb);
  proj_k <<<1024, 128, 0, stream>>>(KVbf, WkeT, Kb);
  proj_v <<<1024, 128, 0, stream>>>(KVbf, WvT,  Vt);
  proj_kr<<<128,  128, 0, stream>>>(relbf, WkrT, krb);

  // 4. fused attention (WMMA + rel_shift + softmax + WMMA)
  dim3 ag(B_SZ * NHEAD, S_LEN / 16);
  attention_kernel<<<ag, 128, 16 * K_LEN * sizeof(float), stream>>>(
      qu, qvb, Kb, Vt, krb, AO);

  // 5. new_mem = memories, then overwrite slice inside final_proj
  copy_mem<<<6291456 / 256, 256, 0, stream>>>(
      (const float4*)mems, (float4*)(out + (size_t)B_SZ * S_LEN * HDN));

  // 6. output projection + bias + memory-slice write (WMMA, 32x64 tiles)
  final_proj<<<512, 128, 0, stream>>>(AO, WoT, bo, lp, sp, out);
}
